// TFT5Attention_4758823764411
// MI455X (gfx1250) — matrix operand order repaired and checked by reading, not compiled, not run on hardware
//
#include <hip/hip_runtime.h>
#include <hip/hip_bf16.h>

// ---------------------------------------------------------------------------
// T5 encoder self-attention for gfx1250 (MI455X), f16 WMMA with f32 accum.
// B=2, S=2048, D=1024, H=16, Dk=64, NUM_BUCKETS=32, MAX_DISTANCE=128.
// ---------------------------------------------------------------------------

typedef __attribute__((ext_vector_type(16))) _Float16 v16h;
typedef __attribute__((ext_vector_type(8)))  _Float16 v8h;
typedef __attribute__((ext_vector_type(8)))  float    v8f;

#define WMMA_F16(a, b, c) \
  __builtin_amdgcn_wmma_f32_16x16x32_f16(false, (a), false, (b), (short)0, (c), false, false)

constexpr int Bn = 2;
constexpr int Sn = 2048;
constexpr int Dn = 1024;
constexpr int Hn = 16;
constexpr int Dk = 64;

// ---------------------------------------------------------------------------
// Fragment loaders (contiguous k order k = 16*(lane/16) + s, same for A and B)
// ---------------------------------------------------------------------------

// A-fragment, 16x32 f16, from row-major src (lda in elements).
// lane&15 = row M; lanes 0-15 hold K k0+0..15; lanes 16-31 hold K k0+16..31.
__device__ inline v16h load_a_frag(const _Float16* __restrict__ src, int lda,
                                   int row0, int k0, int lane) {
  int r = lane & 15;
  int koff = (lane >> 4) ? 16 : 0;
  const _Float16* p = src + (size_t)(row0 + r) * lda + k0 + koff;
  return *(const v16h*)p;           // K = k0+koff .. +15  (32B contiguous)
}

// Same A-fragment layout but sourced from LDS (contiguous f16 rows, stride lds).
__device__ inline v16h load_a_frag_lds16(const _Float16* pc, int lds,
                                         int k0, int lane) {
  int r = lane & 15;
  int koff = (lane >> 4) ? 16 : 0;
  const _Float16* base = pc + (size_t)r * lds + k0 + koff;
  return *(const v16h*)base;          // K = k0+koff .. +15
}

// B-fragment, 32x16 f16, from "column-major" Bt where Bt[n*ldb + k] = B[k][n].
// lane&15 = col N; lanes 0-15 hold K k0+0..15; lanes 16-31 hold K k0+16..31.
__device__ inline v16h load_b_frag(const _Float16* __restrict__ Bt, int ldb,
                                   int n0, int k0, int lane) {
  int n = lane & 15;
  int koff = (lane >> 4) ? 16 : 0;
  return *(const v16h*)(Bt + (size_t)(n0 + n) * ldb + k0 + koff);  // 32B contiguous
}

// T5 relative position bucket for n = q - k (bidirectional, 32 buckets, max 128)
__device__ inline int t5_bucket(int qmk) {
  int ret = (qmk < 0) ? 16 : 0;
  int an = (qmk < 0) ? -qmk : qmk;
  int bkt;
  if (an < 8) {
    bkt = an;
  } else {
    // 8 + int( log(an/8) / log(16) * 8 ),  8/log(16) = 2.88539008...
    int v = 8 + (int)(__logf((float)an * 0.125f) * 2.8853900817779268f);
    bkt = v < 15 ? v : 15;
  }
  return ret + bkt;
}

// ---------------------------------------------------------------------------
// Packing kernels
// ---------------------------------------------------------------------------
__global__ void cast_f32_to_f16(const float* __restrict__ in,
                                _Float16* __restrict__ out, int n) {
  int i = blockIdx.x * blockDim.x + threadIdx.x;
  if (i < n) out[i] = (_Float16)in[i];
}

// in: [rows][cols] f32 row-major  ->  out: [cols][rows] f16 ("column-major" pack)
__global__ void transpose_to_f16(const float* __restrict__ in,
                                 _Float16* __restrict__ out, int rows, int cols) {
  int i = blockIdx.x * blockDim.x + threadIdx.x;
  if (i < rows * cols) {
    int r = i / cols, c = i - r * cols;
    out[(size_t)c * rows + r] = (_Float16)in[i];
  }
}

// ---------------------------------------------------------------------------
// Generic WMMA GEMM: C[M,N] = A[M,K] * Bt[N,K]^T
// Block = 256 threads (8 waves); block tile 128(M) x 64(N); wave tile 16 x 64.
// MODE 0: f32 out, row-major [M,N]
// MODE 1: f16 out, head-split [B,H,S,Dk]   (for Q, K)
// MODE 2: f16 out, transposed [B,H,Dk,S]   (for V)
// ---------------------------------------------------------------------------
template <int MODE>
__global__ __launch_bounds__(256) void gemm_wmma(const _Float16* __restrict__ A,
                                                 const _Float16* __restrict__ Bt,
                                                 void* __restrict__ out,
                                                 int M, int N, int Kd) {
  int lane = threadIdx.x & 31;
  int wave = threadIdx.x >> 5;
  int m0 = blockIdx.x * 128 + wave * 16;
  int n0 = blockIdx.y * 64;

  v8f c[4];
#pragma unroll
  for (int t = 0; t < 4; ++t) c[t] = {};

  for (int kk = 0; kk < Kd; kk += 32) {
    if (kk + 32 < Kd)  // hint next A line into cache (global_prefetch_b8)
      __builtin_prefetch(A + (size_t)(m0 + (lane & 15)) * Kd + kk + 32, 0, 1);
    v16h a = load_a_frag(A, Kd, m0, kk, lane);
#pragma unroll
    for (int t = 0; t < 4; ++t) {
      v16h b = load_b_frag(Bt, Kd, n0 + t * 16, kk, lane);
      c[t] = WMMA_F16(a, b, c[t]);
    }
  }

  int rbase = (lane >> 4) ? 8 : 0;
  int col_l = lane & 15;
#pragma unroll
  for (int t = 0; t < 4; ++t) {
#pragma unroll
    for (int r = 0; r < 8; ++r) {
      int row = m0 + rbase + r;
      int col = n0 + t * 16 + col_l;
      float v = c[t][r];
      if (MODE == 0) {
        ((float*)out)[(size_t)row * N + col] = v;
      } else if (MODE == 1) {
        int b = row >> 11, s = row & (Sn - 1);
        int h = col >> 6,  d = col & (Dk - 1);
        ((_Float16*)out)[(((size_t)(b * Hn + h) * Sn + s) << 6) + d] = (_Float16)v;
      } else {
        int b = row >> 11, s = row & (Sn - 1);
        int h = col >> 6,  d = col & (Dk - 1);
        ((_Float16*)out)[(((size_t)(b * Hn + h) * Dk + d) << 11) + s] = (_Float16)v;
      }
    }
  }
}

// ---------------------------------------------------------------------------
// Attention kernel: one block per (b, h, 16-query tile). 256 threads, 8 waves.
// Dynamic LDS: 16 x 2048 f32 score strip (128 KB) + 16 x 2048 f16 P (64 KB).
// Static LDS: 2064-entry bias LUT (one entry per relative distance) + scratch.
// ---------------------------------------------------------------------------
__global__ __launch_bounds__(256) void attn_kernel(const _Float16* __restrict__ Q,
                                                   const _Float16* __restrict__ Kh,
                                                   const _Float16* __restrict__ Vt,
                                                   const float* __restrict__ rel_emb,
                                                   _Float16* __restrict__ ctx) {
  extern __shared__ float smem[];          // [16][Sn] f32 scores, then f16 P
  __shared__ float biasLUT[2064];          // bias vs (q - k), idx = q-k-(q0-2047)
  __shared__ float red[16][17];            // per-row partial reductions

  int lane = threadIdx.x & 31;
  int wave = threadIdx.x >> 5;
  int qt = blockIdx.x & (Sn / 16 - 1);     // 0..127
  int h  = (blockIdx.x >> 7) & (Hn - 1);
  int b  = blockIdx.x >> 11;
  int q0 = qt * 16;

  // Bias LUT: distance q-k spans [q0-2047, q0+15] -> 2063 entries.
  for (int idx = threadIdx.x; idx < 2063; idx += 256) {
    biasLUT[idx] = rel_emb[t5_bucket(q0 - 2047 + idx) * Hn + h];
  }
  __syncthreads();

  const _Float16* Qh  = Q  + ((size_t)(b * Hn + h) * Sn) * Dk;   // [S][Dk]
  const _Float16* Khd = Kh + ((size_t)(b * Hn + h) * Sn) * Dk;   // [S][Dk]
  const _Float16* Vhd = Vt + ((size_t)(b * Hn + h) * Dk) * Sn;   // [Dk][S]

  int rbase = (lane >> 4) ? 8 : 0;
  int col_l = lane & 15;

  // ---- Phase 1: scores = Q K^T + bias, written f32 into LDS --------------
  v16h a0 = load_a_frag(Qh, Dk, q0, 0, lane);
  v16h a1 = load_a_frag(Qh, Dk, q0, 32, lane);
  for (int nt = wave; nt < Sn / 16; nt += 8) {   // uniform per wave
    int k0 = nt * 16;
    v8f c = {};
    v16h b0 = load_b_frag(Khd, Dk, k0, 0, lane);   // B = K^T (lane = key)
    c = WMMA_F16(a0, b0, c);
    v16h b1 = load_b_frag(Khd, Dk, k0, 32, lane);
    c = WMMA_F16(a1, b1, c);
    int kpos = k0 + col_l;
#pragma unroll
    for (int r = 0; r < 8; ++r) {
      int row = rbase + r;
      smem[row * Sn + kpos] = c[r] + biasLUT[row - kpos + 2047];
    }
  }
  __syncthreads();

  // ---- Phase 2: softmax over each of 16 rows (16 threads per row) --------
  int r   = threadIdx.x >> 4;   // 0..15
  int seg = threadIdx.x & 15;   // 0..15, each owns 128 contiguous elements
  float* rowp = smem + (size_t)r * Sn;
  _Float16* pc = (_Float16*)(smem + 16 * Sn);   // compact f16 P [16][Sn]

  float m = -3.0e30f;
  for (int k = seg * 128; k < seg * 128 + 128; ++k) m = fmaxf(m, rowp[k]);
  red[r][seg] = m;
  __syncthreads();
  m = red[r][0];
#pragma unroll
  for (int i = 1; i < 16; ++i) m = fmaxf(m, red[r][i]);
  __syncthreads();

  float sum = 0.0f;
  for (int k = seg * 128; k < seg * 128 + 128; ++k) {
    float e = __expf(rowp[k] - m);
    rowp[k] = e;
    sum += e;
  }
  red[r][seg] = sum;
  __syncthreads();
  sum = 0.0f;
#pragma unroll
  for (int i = 0; i < 16; ++i) sum += red[r][i];
  float inv = 1.0f / sum;

  // normalize; pack f16 contiguously into the separate P region (reads from
  // the f32 region, writes to the f16 region -> no aliasing hazards)
  for (int k = seg * 128; k < seg * 128 + 128; ++k) {
    pc[(size_t)r * Sn + k] = (_Float16)(rowp[k] * inv);
  }
  __syncthreads();

  // ---- Phase 3: context = P V (waves 0..3 each own one 16-wide d tile) ---
  if (wave < 4) {                          // uniform per wave -> EXEC all ones
    int d0 = wave * 16;
    v8f c = {};
    for (int kk = 0; kk < Sn; kk += 32) {
      v16h ap = load_a_frag_lds16(pc, Sn, kk, lane);  // 2x ds_load_b128
      v16h bp = load_b_frag(Vhd, Sn, d0, kk, lane);   // Vt: [d][key]
      c = WMMA_F16(ap, bp, c);
    }
#pragma unroll
    for (int rr = 0; rr < 8; ++rr) {
      int row = rbase + rr;
      int d   = d0 + col_l;
      ctx[(((size_t)(b * Sn + q0 + row)) << 10) + h * Dk + d] = (_Float16)c[rr];
    }
  }
}

// ---------------------------------------------------------------------------
// Host-side launch
// ---------------------------------------------------------------------------
extern "C" void kernel_launch(void* const* d_in, const int* in_sizes, int n_in,
                              void* d_out, int out_size, void* d_ws, size_t ws_size,
                              hipStream_t stream) {
  (void)in_sizes; (void)n_in; (void)out_size; (void)ws_size;

  const float* x   = (const float*)d_in[0];   // [B,S,D]
  const float* Wq  = (const float*)d_in[1];   // [D, H*Dk]
  const float* Wk  = (const float*)d_in[2];
  const float* Wv  = (const float*)d_in[3];
  const float* Wo  = (const float*)d_in[4];   // [H*Dk, D]
  const float* rel = (const float*)d_in[5];   // [32, H]
  float* out = (float*)d_out;                 // [B,S,D] f32

  // Workspace layout (48 MiB total)
  char* ws = (char*)d_ws;
  _Float16* xh  = (_Float16*)(ws);                       //  8 MiB  x as f16
  _Float16* WqT = (_Float16*)(ws + ( 8ull << 20));       //  2 MiB  Wq^T f16
  _Float16* WkT = (_Float16*)(ws + (10ull << 20));       //  2 MiB
  _Float16* WvT = (_Float16*)(ws + (12ull << 20));       //  2 MiB
  _Float16* WoT = (_Float16*)(ws + (14ull << 20));       //  2 MiB
  _Float16* Qh  = (_Float16*)(ws + (16ull << 20));       //  8 MiB  [B,H,S,Dk]
  _Float16* Kh  = (_Float16*)(ws + (24ull << 20));       //  8 MiB  [B,H,S,Dk]
  _Float16* Vt  = (_Float16*)(ws + (32ull << 20));       //  8 MiB  [B,H,Dk,S]
  _Float16* ctx = (_Float16*)(ws + (40ull << 20));       //  8 MiB  [B,S,H*Dk]

  const int M = Bn * Sn;        // 4096
  const int nX = M * Dn;        // 4,194,304

  cast_f32_to_f16<<<(nX + 255) / 256, 256, 0, stream>>>(x, xh, nX);
  transpose_to_f16<<<(Dn * Dn + 255) / 256, 256, 0, stream>>>(Wq, WqT, Dn, Hn * Dk);
  transpose_to_f16<<<(Dn * Dn + 255) / 256, 256, 0, stream>>>(Wk, WkT, Dn, Hn * Dk);
  transpose_to_f16<<<(Dn * Dn + 255) / 256, 256, 0, stream>>>(Wv, WvT, Dn, Hn * Dk);
  transpose_to_f16<<<(Dn * Dn + 255) / 256, 256, 0, stream>>>(Wo, WoT, Hn * Dk, Dn);

  dim3 gg(M / 128, Dn / 64);    // 32 x 16
  dim3 bb(256);
  gemm_wmma<1><<<gg, bb, 0, stream>>>(xh, WqT, Qh, M, Dn, Dn);
  gemm_wmma<1><<<gg, bb, 0, stream>>>(xh, WkT, Kh, M, Dn, Dn);
  gemm_wmma<2><<<gg, bb, 0, stream>>>(xh, WvT, Vt, M, Dn, Dn);

  // 128 KiB f32 scores + 64 KiB f16 probabilities (WGP LDS = 320 KiB)
  size_t attn_lds = (size_t)16 * Sn * sizeof(float) + (size_t)16 * Sn * sizeof(_Float16);
  attn_kernel<<<Bn * Hn * (Sn / 16), 256, attn_lds, stream>>>(Qh, Kh, Vt, rel, ctx);

  gemm_wmma<0><<<gg, bb, 0, stream>>>(ctx, WoT, out, M, Dn, Dn);
}
